// TreeLSTMCell_26534307955066
// MI455X (gfx1250) — matrix operand order repaired and checked by reading, not compiled, not run on hardware
//
#include <hip/hip_runtime.h>

// ---------------------------------------------------------------------------
// TreeLSTM cell, fused, MI455X (gfx1250, wave32, WMMA bf16 16x16x32).
//
// Roofline: 64.4 GFLOP, ~610 MB unavoidable HBM traffic -> ~26us @ 23.3 TB/s;
// bf16 WMMA compute is comparable, so GEMM operands are bf16 (f32 accum).
// Three kernels:
//   K0: f_input = x@W_fin + b_fin        -> ws (packed bf16 C-tiles, 16 MB)
//   K1: forget path (W_f from LDS): f = sigmoid(nh@W_f + f_input + b_f),
//       c_aggr = sum_c f*nc, h_sum accumulated in bf16 A-fragments -> ws
//   K2: iou = x@W_iou + h_sum@W_aggr + biases; nonlinear epilogue.
// Round-2 spilled because fi[16] (128 f32 VGPRs) lived across K1's child
// loop; moving the GEMM to K0 and keeping fi packed-bf16 (64 VGPRs) brings
// K1 to ~380 VGPRs peak. K0/K2 are light so occupancy hides the L2 latency
// of streamed weight fragments.
// ---------------------------------------------------------------------------

typedef __attribute__((ext_vector_type(16))) __bf16 v16bf;
typedef __attribute__((ext_vector_type(8)))  __bf16 v8bf;
typedef __attribute__((ext_vector_type(8)))  float  v8f;
typedef __attribute__((ext_vector_type(4)))  float  v4f;

#define N_NODES 32768
#define CCH     8
#define HS      256
#define XS      256
#define PITCH   264   // 256 bf16 + 8 pad = 528 B rows -> conflict-free LDS banks
#define WAVES0  8
#define WAVES1  4     // forget kernel: 4 waves (128 thr), 132 KB LDS
#define WAVES2  8
#define NODES_PER_WAVE 16
#define NTILES  (N_NODES / NODES_PER_WAVE)   // 2048 wave-tiles

// ws layout (bytes)
#define WS_WTF    0                          // 256*264*2
#define WS_WTFIN  (256 * PITCH * 2)
#define WS_WTIOU  (2 * 256 * PITCH * 2)
#define WS_WTAGGR (2 * 256 * PITCH * 2 + 768 * PITCH * 2)
#define WS_CA     (2 * 256 * PITCH * 2 + 2 * 768 * PITCH * 2)   // f32 [N][256]
#define WS_HSUM   (WS_CA + N_NODES * HS * 4)                     // bf16 frag-linear
#define WS_FI     (WS_HSUM + N_NODES * HS * 2)                   // bf16 packed C-tiles

// ---------------------------------------------------------------------------
// Prologue: W [256 x O] f32 (row = input k) -> WT [O x PITCH] bf16 (row = out col)
// ---------------------------------------------------------------------------
__global__ void transpose_bf16_kernel(const float* __restrict__ W,
                                      unsigned short* __restrict__ WT_u, int O) {
    __bf16* WT = (__bf16*)WT_u;
    int idx = blockIdx.x * blockDim.x + threadIdx.x;
    int total = O * PITCH;
    if (idx >= total) return;
    int o = idx / PITCH;
    int i = idx - o * PITCH;
    float v = (i < XS) ? W[i * O + o] : 0.0f;
    WT[idx] = (__bf16)v;
}

// ---------------------------------------------------------------------------
// Device helpers
// ---------------------------------------------------------------------------
__device__ __forceinline__ float sigmoidf_(float x) {
    return 1.0f / (1.0f + __expf(-x));
}
__device__ __forceinline__ float tanhf_(float x) {
    float e = __expf(2.0f * x);
    return 1.0f - 2.0f / (e + 1.0f);   // exact at +-inf, no NaN
}
__device__ __forceinline__ unsigned pack_bf16(float a, float b) {
    union { __bf16 h[2]; unsigned u; } v;
    v.h[0] = (__bf16)a; v.h[1] = (__bf16)b;
    return v.u;                         // -> v_cvt_pk_bf16_f32
}
__device__ __forceinline__ float unpack_bf16(unsigned u, int h) {
    union { unsigned u; __bf16 b[2]; } v;
    v.u = u;
    return (float)v.b[h];
}

// A-fragment (16x32 bf16, M on lanes): lane lo holds row M=lo,
// elems 0..7 = K c0..c0+7, elems 8..15 = K c0+16..c0+23, c0 = 32k + 8*hi.
__device__ __forceinline__ v16bf load_a_frag(const float* __restrict__ row,
                                             int k, int hi) {
    int c0 = k * 32 + hi * 8;
    v4f a = *(const v4f*)(row + c0);
    v4f b = *(const v4f*)(row + c0 + 4);
    v4f c = *(const v4f*)(row + c0 + 16);
    v4f d = *(const v4f*)(row + c0 + 20);
    v16bf r;
#pragma unroll
    for (int t = 0; t < 4; ++t) {
        r[t]      = (__bf16)a[t];
        r[4 + t]  = (__bf16)b[t];
        r[8 + t]  = (__bf16)c[t];
        r[12 + t] = (__bf16)d[t];
    }
    return r;
}

// Same, but also accumulates the loaded f32 values into a bf16 h_sum fragment
// (bf16 storage: the sum is rounded to bf16 for the WMMA anyway; saves 64 VGPRs).
__device__ __forceinline__ v16bf load_a_frag_acc(const float* __restrict__ row,
                                                 int k, int hi, v16bf& hsum) {
    int c0 = k * 32 + hi * 8;
    v4f a = *(const v4f*)(row + c0);
    v4f b = *(const v4f*)(row + c0 + 4);
    v4f c = *(const v4f*)(row + c0 + 16);
    v4f d = *(const v4f*)(row + c0 + 20);
    v16bf r;
#pragma unroll
    for (int t = 0; t < 4; ++t) {
        r[t]      = (__bf16)a[t];
        r[4 + t]  = (__bf16)b[t];
        r[8 + t]  = (__bf16)c[t];
        r[12 + t] = (__bf16)d[t];
        hsum[t]      = (__bf16)((float)hsum[t]      + a[t]);
        hsum[4 + t]  = (__bf16)((float)hsum[4 + t]  + b[t]);
        hsum[8 + t]  = (__bf16)((float)hsum[8 + t]  + c[t]);
        hsum[12 + t] = (__bf16)((float)hsum[12 + t] + d[t]);
    }
    return r;
}

// B-fragment (32x16 bf16, N on lanes): lane holds column N = jt*16+lo,
// elems 0..7 = K c0..c0+7, elems 8..15 = K c0+16..c0+23, c0 = 32k + 8*hi.
__device__ __forceinline__ v16bf load_b_frag(const __bf16* M, int jt, int k,
                                             int lo, int hi) {
    const __bf16* p = M + (jt * 16 + lo) * PITCH + k * 32 + hi * 8;
    union { v16bf v; v8bf h[2]; } u;
    u.h[0] = *(const v8bf*)p;
    u.h[1] = *(const v8bf*)(p + 16);
    return u.v;
}

#define WMMA_BF16(A, B, C) \
    __builtin_amdgcn_wmma_f32_16x16x32_bf16(false, (A), false, (B), (short)0, (C), false, false)

// ---------------------------------------------------------------------------
// K0: f_input = x @ W_fin + b_fin, stored as packed-bf16 C-tiles:
// per tile T, per j, per lane: uint4 = rows {0,1|2,3|4,5|6,7} pair-packed.
// ---------------------------------------------------------------------------
__global__ void
__launch_bounds__(WAVES0 * 32)
treelstm_fin_kernel(const float* __restrict__ x,
                    const float* __restrict__ b_fin,
                    const unsigned short* __restrict__ wtfin_u,
                    unsigned short* __restrict__ fi_ws) {
    const __bf16* wtfin = (const __bf16*)wtfin_u;

    const int wave = threadIdx.x >> 5;
    const int lane = threadIdx.x & 31;
    const int lo = lane & 15;
    const int hi = lane >> 4;
    const int T = blockIdx.x * WAVES0 + wave;
    const int base = T * NODES_PER_WAVE;

    v16bf xf[8];
    const float* xrow = x + (base + lo) * XS;
#pragma unroll
    for (int k = 0; k < 8; ++k) xf[k] = load_a_frag(xrow, k, hi);

#pragma unroll
    for (int j = 0; j < 16; ++j) {
        v8f acc = {};
#pragma unroll
        for (int k = 0; k < 8; ++k)
            acc = WMMA_BF16(xf[k], load_b_frag(wtfin, j, k, lo, hi), acc);
        float bv = b_fin[j * 16 + lo];
        uint4 p;
        p.x = pack_bf16(acc[0] + bv, acc[1] + bv);
        p.y = pack_bf16(acc[2] + bv, acc[3] + bv);
        p.z = pack_bf16(acc[4] + bv, acc[5] + bv);
        p.w = pack_bf16(acc[6] + bv, acc[7] + bv);
        *(uint4*)(fi_ws + ((T * 16 + j) * 32 + lane) * 8) = p;
    }
}

// ---------------------------------------------------------------------------
// K1 (forget path): per child f = sigmoid(nh@W_f + f_input + b_f);
// c_aggr = sum_c f*nc; h_sum accumulated in bf16 A-fragments.
// W_f^T lives in LDS; f_input held as packed bf16 (64 VGPRs).
// ---------------------------------------------------------------------------
__global__ void
__launch_bounds__(WAVES1 * 32, 1)
__attribute__((amdgpu_waves_per_eu(1)))
treelstm_forget_kernel(const float* __restrict__ nh,
                       const float* __restrict__ nc,
                       const float* __restrict__ b_f,
                       const unsigned short* __restrict__ wtf_u,
                       const unsigned short* __restrict__ fi_ws,
                       float* __restrict__ ca_ws,
                       unsigned short* __restrict__ hs_ws) {
    // Stage W_f^T (bf16, padded rows) into LDS: 256*264*2 = 132 KB.
    __shared__ __bf16 sWf[HS * PITCH];
    {
        const uint4* src = (const uint4*)wtf_u;
        uint4* dst = (uint4*)sWf;
        const int n16 = HS * PITCH * 2 / 16;
        for (int i = threadIdx.x; i < n16; i += WAVES1 * 32) dst[i] = src[i];
    }
    __syncthreads();

    const int wave = threadIdx.x >> 5;
    const int lane = threadIdx.x & 31;
    const int lo = lane & 15;
    const int hi = lane >> 4;
    const int T = blockIdx.x * WAVES1 + wave;
    const int base = T * NODES_PER_WAVE;

    // f_input tiles, packed bf16: fib[j][r>>1] halves = rows r, r+1.
    unsigned fib[16][4];
#pragma unroll
    for (int j = 0; j < 16; ++j) {
        uint4 t = *(const uint4*)(fi_ws + ((T * 16 + j) * 32 + lane) * 8);
        fib[j][0] = t.x; fib[j][1] = t.y; fib[j][2] = t.z; fib[j][3] = t.w;
    }

    v8f ca[16];
#pragma unroll
    for (int j = 0; j < 16; ++j) ca[j] = (v8f){};
    v16bf hsf[8];
#pragma unroll
    for (int k = 0; k < 8; ++k) {
#pragma unroll
        for (int e = 0; e < 16; ++e) hsf[k][e] = (__bf16)0.0f;
    }

#pragma unroll 1
    for (int c = 0; c < CCH; ++c) {
        v16bf nhf[8];
        const float* hrow = nh + ((base + lo) * CCH + c) * HS;
#pragma unroll
        for (int k = 0; k < 8; ++k) nhf[k] = load_a_frag_acc(hrow, k, hi, hsf[k]);

#pragma unroll
        for (int j = 0; j < 16; ++j) {
            v8f g = {};
#pragma unroll
            for (int k = 0; k < 8; ++k)
                g = WMMA_BF16(nhf[k], load_b_frag(sWf, j, k, lo, hi), g);
            float bfv = b_f[j * 16 + lo];
            const float* ncp = nc + ((base + 8 * hi) * CCH + c) * HS + j * 16 + lo;
#pragma unroll
            for (int r = 0; r < 8; ++r) {
                float fiv = unpack_bf16(fib[j][r >> 1], r & 1);
                float fz  = sigmoidf_(g[r] + bfv + fiv);
                float ncv = ncp[r * (CCH * HS)];
                ca[j][r] += fz * ncv;
            }
        }
    }

    // hand-off: c_aggr (row-major f32) and h_sum frags (lane-linear bf16)
#pragma unroll
    for (int j = 0; j < 16; ++j) {
        int col = j * 16 + lo;
#pragma unroll
        for (int r = 0; r < 8; ++r)
            ca_ws[(base + r + 8 * hi) * HS + col] = ca[j][r];
    }
#pragma unroll
    for (int k = 0; k < 8; ++k)
        *(v16bf*)((__bf16*)hs_ws + (T * 8 + k) * 512 + lane * 16) = hsf[k];
}

// ---------------------------------------------------------------------------
// K2 (iou path): iou = x@W_iou + h_sum@W_aggr + biases; nonlinear epilogue
// with c_aggr from ws. Light registers -> occupancy hides L2 streaming.
// ---------------------------------------------------------------------------
__global__ void
__launch_bounds__(WAVES2 * 32)
treelstm_iou_kernel(const float* __restrict__ x,
                    const float* __restrict__ b_iou,
                    const float* __restrict__ b_aggr,
                    const unsigned short* __restrict__ wtiou_u,
                    const unsigned short* __restrict__ wtaggr_u,
                    const float* __restrict__ ca_ws,
                    const unsigned short* __restrict__ hs_ws,
                    float* __restrict__ out) {
    const __bf16* wtiou  = (const __bf16*)wtiou_u;
    const __bf16* wtaggr = (const __bf16*)wtaggr_u;

    const int wave = threadIdx.x >> 5;
    const int lane = threadIdx.x & 31;
    const int lo = lane & 15;
    const int hi = lane >> 4;
    const int T = blockIdx.x * WAVES2 + wave;
    const int base = T * NODES_PER_WAVE;

    v16bf xf[8], hsf[8];
    const float* xrow = x + (base + lo) * XS;
#pragma unroll
    for (int k = 0; k < 8; ++k) {
        xf[k]  = load_a_frag(xrow, k, hi);
        hsf[k] = *(const v16bf*)((const __bf16*)hs_ws + (T * 8 + k) * 512 + lane * 16);
    }

    float* __restrict__ outH = out;
    float* __restrict__ outC = out + N_NODES * HS;

#pragma unroll
    for (int j = 0; j < 16; ++j) {
        v8f ai = {}, ao = {}, au = {};
#pragma unroll
        for (int k = 0; k < 8; ++k) {
            ai = WMMA_BF16(xf[k],  load_b_frag(wtiou,       j, k, lo, hi), ai);
            ao = WMMA_BF16(xf[k],  load_b_frag(wtiou,  16 + j, k, lo, hi), ao);
            au = WMMA_BF16(xf[k],  load_b_frag(wtiou,  32 + j, k, lo, hi), au);
            ai = WMMA_BF16(hsf[k], load_b_frag(wtaggr,      j, k, lo, hi), ai);
            ao = WMMA_BF16(hsf[k], load_b_frag(wtaggr, 16 + j, k, lo, hi), ao);
            au = WMMA_BF16(hsf[k], load_b_frag(wtaggr, 32 + j, k, lo, hi), au);
        }
        int col = j * 16 + lo;
        float bi = b_iou[col]       + b_aggr[col];
        float bo = b_iou[256 + col] + b_aggr[256 + col];
        float bu = b_iou[512 + col] + b_aggr[512 + col];
#pragma unroll
        for (int r = 0; r < 8; ++r) {
            int row = base + r + 8 * hi;
            float iv = sigmoidf_(ai[r] + bi);
            float uv = tanhf_(au[r] + bu);
            float cn = iv * uv + ca_ws[row * HS + col];
            float ov = sigmoidf_(ao[r] + bo);
            float hn = ov * tanhf_(cn);
            outH[row * HS + col] = hn;
            outC[row * HS + col] = cn;
        }
    }
}

// ---------------------------------------------------------------------------
// Host launch
// ---------------------------------------------------------------------------
extern "C" void kernel_launch(void* const* d_in, const int* in_sizes, int n_in,
                              void* d_out, int out_size, void* d_ws, size_t ws_size,
                              hipStream_t stream) {
    const float* x      = (const float*)d_in[0];
    const float* nh     = (const float*)d_in[1];
    const float* nc     = (const float*)d_in[2];
    const float* W_iou  = (const float*)d_in[3];
    const float* b_iou  = (const float*)d_in[4];
    const float* W_fin  = (const float*)d_in[5];
    const float* b_fin  = (const float*)d_in[6];
    const float* W_f    = (const float*)d_in[7];
    const float* b_f    = (const float*)d_in[8];
    const float* W_aggr = (const float*)d_in[9];
    const float* b_aggr = (const float*)d_in[10];
    float* out = (float*)d_out;

    char* ws = (char*)d_ws;
    unsigned short* wtf    = (unsigned short*)(ws + WS_WTF);
    unsigned short* wtfin  = (unsigned short*)(ws + WS_WTFIN);
    unsigned short* wtiou  = (unsigned short*)(ws + WS_WTIOU);
    unsigned short* wtaggr = (unsigned short*)(ws + WS_WTAGGR);
    float*          ca_ws  = (float*)(ws + WS_CA);
    unsigned short* hs_ws  = (unsigned short*)(ws + WS_HSUM);
    unsigned short* fi_ws  = (unsigned short*)(ws + WS_FI);

    {
        int t = 256 * PITCH;
        transpose_bf16_kernel<<<(t + 255) / 256, 256, 0, stream>>>(W_f,   wtf,   256);
        transpose_bf16_kernel<<<(t + 255) / 256, 256, 0, stream>>>(W_fin, wtfin, 256);
        t = 768 * PITCH;
        transpose_bf16_kernel<<<(t + 255) / 256, 256, 0, stream>>>(W_iou,  wtiou,  768);
        transpose_bf16_kernel<<<(t + 255) / 256, 256, 0, stream>>>(W_aggr, wtaggr, 768);
    }

    dim3 grid0(NTILES / WAVES0);   // 256 blocks x 256 thr
    treelstm_fin_kernel<<<grid0, WAVES0 * 32, 0, stream>>>(x, b_fin, wtfin, fi_ws);

    dim3 grid1(NTILES / WAVES1);   // 512 blocks x 128 thr
    treelstm_forget_kernel<<<grid1, WAVES1 * 32, 0, stream>>>(
        nh, nc, b_f, wtf, fi_ws, ca_ws, hs_ws);

    dim3 grid2(NTILES / WAVES2);   // 256 blocks x 256 thr
    treelstm_iou_kernel<<<grid2, WAVES2 * 32, 0, stream>>>(
        x, b_iou, b_aggr, wtiou, wtaggr, ca_ws, hs_ws, out);
}
